// SparseQRNNLayer_18159121727992
// MI455X (gfx1250) — matrix operand order repaired and checked by reading, not compiled, not run on hardware
//
#include <hip/hip_runtime.h>
#include <hip/hip_bf16.h>
#include <math.h>

typedef __attribute__((ext_vector_type(16))) _Float16 v16h;
typedef __attribute__((ext_vector_type(8)))  float    v8f;

#define NELEM   33554432UL   // 4*16*32*128*128 (one gate tensor)
#define KREAL   432          // 16 ch * 27 taps
#define KPAD    448          // 14 * 32
#define NKB     14
#define PSTRIDE 464          // padded k-stride in halves (mult of 16 for 32B-aligned v16h loads)
#define NROW    144          // 16 ch * 9 (dt,dh) rows
#define NU      130          // input W positions -1..128 per row

// ---------------------------------------------------------------------------
// Kernel 1: Conv3d(16->32, 3x3x3, pad=1) as implicit GEMM with WMMA f16->f32.
// One workgroup per (b, t, h) row of W=128. 8 waves, each owns a 16-wide W
// tile and computes a 32(ch) x 16(w) output tile over K=432 (padded 448).
// Writes Z = tanh(gate[0:16]) to out[0..N) and F = sigmoid(gate[16:32]) to
// out[N..2N)  (d_out doubles as scratch for kernel 2).
// ---------------------------------------------------------------------------
__global__ __launch_bounds__(256)
void qrnn_conv_wmma(const float* __restrict__ x,
                    const float* __restrict__ cw,
                    const float* __restrict__ cb,
                    float* __restrict__ out) {
  __shared__ __attribute__((aligned(32))) _Float16 sPatch[128 * PSTRIDE];   // B: [w][k]
  __shared__ __attribute__((aligned(32))) _Float16 sA[2 * NKB * 32 * 16];   // A fragments

  const int tid = threadIdx.x;
  const int bid = blockIdx.x;
  const int h = bid & 127;
  const int t = (bid >> 7) & 31;
  const int b = bid >> 12;

  // ---- Stage weights pre-swizzled into exact A-fragment VGPR layout ----
  // A 16x32 fragment, contiguous k order used here (B matches): lanes 0-15 row M=lane hold K in {0..15};
  // lanes 16-31 row M=lane-16 hold K in {16..31}. Element e of v16h:
  // K = e + (lane&16 ? 16 : 0).
  for (int idx = tid; idx < 2 * NKB * 32 * 16; idx += 256) {
    const int e    = idx & 15;
    const int lane = (idx >> 4) & 31;
    const int g    = idx >> 9;           // mt*NKB + kb
    const int kb   = g % NKB;
    const int mt   = g / NKB;
    const int kl   = e + ((lane & 16) ? 16 : 0);
    const int k    = kb * 32 + kl;
    const int o    = mt * 16 + (lane & 15);      // output channel (A row)
    _Float16 v = (_Float16)0.f;
    if (k < KREAL) v = (_Float16)cw[o * KREAL + k];   // k == flat (i,dt,dh,kw)
    sA[idx] = v;
  }

  // ---- Zero the K-padding band of the patch (k in [KREAL, KPAD)) ----
  for (int idx = tid; idx < 128 * (KPAD - KREAL); idx += 256) {
    const int w = idx >> 4;              // 16 pad slots per w
    const int k = KREAL + (idx & 15);
    sPatch[w * PSTRIDE + k] = (_Float16)0.f;
  }

  // ---- Stage im2col patch, source-major: one global load per (row, u), ----
  // ---- scattered to its <=3 (w, k) destinations in B order sPatch[w][k]. ----
  // Row r = (i*3 + dt)*3 + dh; u = input W index + 1 (0..129).
  // Destination: k = r*3 + kw, w = u - kw  (so x[.., u-1] feeds taps kw=0..2).
  for (int idx = tid; idx < NROW * NU; idx += 256) {
    const int r = idx / NU;
    const int u = idx - r * NU;
    const int i  = r / 9;
    const int rr = r - i * 9;
    const int dt = rr / 3;
    const int dh = rr - dt * 3;
    const int tt = t + dt - 1;
    const int hh = h + dh - 1;
    const int ww = u - 1;                // actual input W coordinate

    _Float16 v = (_Float16)0.f;
    if ((unsigned)tt < 32u && (unsigned)hh < 128u && (unsigned)ww < 128u)
      v = (_Float16)x[((((size_t)b * 16 + i) * 32 + tt) * 128 + hh) * 128 + ww];

    const int kb3 = r * 3;
#pragma unroll
    for (int kw = 0; kw < 3; ++kw) {
      const int w = u - kw;
      if ((unsigned)w < 128u) sPatch[w * PSTRIDE + kb3 + kw] = v;
    }
  }
  __syncthreads();

  const int lane  = tid & 31;
  const int wave  = tid >> 5;
  const int w0    = wave << 4;
  const int n     = w0 + (lane & 15);   // this lane's output column (B col N)
  const int klane = lane & 16;          // B layout: lanes 16-31 hold K+16

  v8f acc0 = {};   // channels 0..15  -> Z gates
  v8f acc1 = {};   // channels 16..31 -> F gates
  const v16h* aF = (const v16h*)sA;
#pragma unroll
  for (int kb = 0; kb < NKB; ++kb) {
    v16h a0 = aF[kb * 32 + lane];
    v16h a1 = aF[(NKB + kb) * 32 + lane];
    v16h bf = *(const v16h*)&sPatch[n * PSTRIDE + kb * 32 + klane];
    acc0 = __builtin_amdgcn_wmma_f32_16x16x32_f16(false, a0, false, bf,
                                                  (short)0, acc0, false, false);
    acc1 = __builtin_amdgcn_wmma_f32_16x16x32_f16(false, a1, false, bf,
                                                  (short)0, acc1, false, false);
  }

  // C/D layout: VGPR v, lanes 0-15 -> (M=v, N=lane); lanes 16-31 -> (M=v+8).
  const int Mb = (lane >> 4) << 3;
  const size_t spatial = (size_t)h * 128 + (size_t)n;
#pragma unroll
  for (int v = 0; v < 8; ++v) {
    const int M = Mb + v;
    float zg = acc0[v] + cb[M];
    float fg = acc1[v] + cb[16 + M];
    zg = tanhf(zg);
    fg = 1.0f / (1.0f + __expf(-fg));
    const size_t zi = (((size_t)b * 16 + M) * 32 + t) * 16384 + spatial;
    out[zi]         = zg;   // Z
    out[NELEM + zi] = fg;   // F
  }
}

// ---------------------------------------------------------------------------
// Kernel 2: bidirectional fo-pool scan over T=32, fully in-place on d_out.
// One thread per (b,c,h,w) column: read its whole Z/F column into registers,
// scan forward + backward, then overwrite with (hsl+hsr) and (1-F)*Z.
// ---------------------------------------------------------------------------
__global__ __launch_bounds__(256)
void qrnn_scan(float* __restrict__ out) {
  const size_t p    = (size_t)blockIdx.x * 256 + threadIdx.x;  // over B*C*H*W
  const size_t bc   = p >> 14;          // b*16 + c
  const size_t hw   = p & 16383;
  const size_t base = bc * (32ul * 16384ul) + hw;

  float z[32], f[32], hl[32];
#pragma unroll
  for (int t = 0; t < 32; ++t) {
    z[t] = out[base + (size_t)t * 16384];
    f[t] = out[NELEM + base + (size_t)t * 16384];
  }

  float hacc = 0.f;
#pragma unroll
  for (int t = 0; t < 32; ++t) {
    hacc = f[t] * hacc + (1.f - f[t]) * z[t];
    hl[t] = hacc;
  }

  hacc = 0.f;
#pragma unroll
  for (int t = 31; t >= 0; --t) {
    hacc = f[t] * hacc + (1.f - f[t]) * z[t];
    out[base + (size_t)t * 16384]         = hl[t] + hacc;        // hsl + hsr
    out[NELEM + base + (size_t)t * 16384] = (1.f - f[t]) * z[t]; // hs
  }
}

extern "C" void kernel_launch(void* const* d_in, const int* in_sizes, int n_in,
                              void* d_out, int out_size, void* d_ws, size_t ws_size,
                              hipStream_t stream) {
  const float* x  = (const float*)d_in[0];   // [4,16,32,128,128]
  const float* cw = (const float*)d_in[1];   // [32,16,3,3,3]
  const float* cb = (const float*)d_in[2];   // [32]
  float* out = (float*)d_out;                // [2, 4,16,32,128,128] concat

  // One workgroup per (b, t, h): 4*32*128 = 16384 blocks.
  qrnn_conv_wmma<<<dim3(16384), dim3(256), 0, stream>>>(x, cw, cb, out);
  // One thread per (b, c, h, w): 4*16*128*128 / 256 = 4096 blocks.
  qrnn_scan<<<dim3(4096), dim3(256), 0, stream>>>(out);
}
